// SPA_69191923138590
// MI455X (gfx1250) — matrix operand order repaired and checked by reading, not compiled, not run on hardware
//
#include <hip/hip_runtime.h>

// ---------------- problem constants ----------------
#define BB   4
#define CC   6
#define MID  3
#define HH   257
#define WW   257
#define KS   7
#define STR  4
#define PADA 3
#define LH   65            // output patches per axis
#define LL   (LH*LH)       // 4225 patches
#define DD   294           // C*49
#define DP   320           // padded d (10 * 32)
#define LP   4256          // padded L (133 * 32, 266 * 16)
#define ROWTILES 265       // ceil(L/16)
#define COLTILES (LP/16)   // 266
#define MTILES   (LP/32)   // 133
#define SCL  10.0f

typedef __bf16 bf16t;
typedef __attribute__((ext_vector_type(16))) __bf16 v16bf;
typedef __attribute__((ext_vector_type(8)))  __bf16 v8bf;
typedef __attribute__((ext_vector_type(8)))  float  v8f;

static __device__ inline float bf2f(bf16t b) {
    unsigned short s = __builtin_bit_cast(unsigned short, b);
    unsigned u = ((unsigned)s) << 16;
    return __builtin_bit_cast(float, u);
}
static __device__ inline bf16t f2bf(float f) {
    unsigned u = __builtin_bit_cast(unsigned, f);
    unsigned r = u + 0x7FFFu + ((u >> 16) & 1u);
    unsigned short s = (unsigned short)(r >> 16);
    return __builtin_bit_cast(bf16t, s);
}
static __device__ inline v16bf joinbf(v8bf lo, v8bf hi) {
    return __builtin_shufflevector(lo, hi, 0,1,2,3,4,5,6,7,8,9,10,11,12,13,14,15);
}
// 16 consecutive bf16 (one half-wave chunk in contiguous k order; unused)
static __device__ inline v16bf ld_frag16(const bf16t* p) {
    v8bf lo = *(const v8bf*)(p);
    v8bf hi = *(const v8bf*)(p + 8);
    return joinbf(lo, hi);
}
// two 8-elt chunks 16 elements apart (A-fragment layout)
static __device__ inline v16bf ld_fragA(const bf16t* p) {
    v8bf lo = *(const v8bf*)(p);
    v8bf hi = *(const v8bf*)(p + 16);
    return joinbf(lo, hi);
}
static __device__ inline float leaky(float x) { return x >= 0.0f ? x : 0.2f * x; }

// ---------------- 1) basic block: reflect-pad conv3x3 + leaky + conv1x1 + leaky ----
__global__ void k_basic_block(const float* __restrict__ x,
                              const float* __restrict__ w1,
                              const float* __restrict__ w2,
                              float* __restrict__ xb) {
    int idx = blockIdx.x * blockDim.x + threadIdx.x;
    if (idx >= BB * HH * WW) return;
    int b = idx / (HH * WW);
    int r = idx % (HH * WW);
    int y = r / WW, xo = r % WW;

    float mid[MID] = {0.f, 0.f, 0.f};
    #pragma unroll
    for (int dy = 0; dy < 3; ++dy) {
        int ry = y - 1 + dy;
        ry = ry < 0 ? -ry : (ry >= HH ? 2 * HH - 2 - ry : ry);
        #pragma unroll
        for (int dx = 0; dx < 3; ++dx) {
            int rx = xo - 1 + dx;
            rx = rx < 0 ? -rx : (rx >= WW ? 2 * WW - 2 - rx : rx);
            for (int c = 0; c < CC; ++c) {
                float v = x[(((size_t)b * CC + c) * HH + ry) * WW + rx];
                #pragma unroll
                for (int m = 0; m < MID; ++m)
                    mid[m] += v * w1[((m * CC + c) * 3 + dy) * 3 + dx];
            }
        }
    }
    #pragma unroll
    for (int m = 0; m < MID; ++m) mid[m] = leaky(mid[m]);
    for (int o = 0; o < CC; ++o) {
        float a = 0.f;
        #pragma unroll
        for (int m = 0; m < MID; ++m) a += mid[m] * w2[o * MID + m];
        xb[(((size_t)b * CC + o) * HH + y) * WW + xo] = leaky(a);
    }
}

// ---------------- 2) pack patches to bf16 Q [B][LP][DP] and QT [B][DP][LP] ----
__global__ void k_pack(const float* __restrict__ xb,
                       bf16t* __restrict__ q,
                       bf16t* __restrict__ qT) {
    size_t idx = (size_t)blockIdx.x * blockDim.x + threadIdx.x;
    if (idx >= (size_t)BB * LP * DP) return;
    int b = (int)(idx / ((size_t)LP * DP));
    int rem = (int)(idx % ((size_t)LP * DP));
    int l = rem / DP, d = rem % DP;
    float v = 0.f;
    if (l < LL && d < DD) {
        int c = d / 49, p = d % 49;
        int i = p / 7, j = p % 7;
        int lh = l / LH, lw = l % LH;
        int y = STR * lh + i - PADA;
        int x = STR * lw + j - PADA;
        if (y >= 0 && y < HH && x >= 0 && x < WW)
            v = xb[(((size_t)b * CC + c) * HH + y) * WW + x];
    }
    bf16t bv = f2bf(v);
    q[((size_t)b * LP + l) * DP + d]  = bv;
    qT[((size_t)b * DP + d) * LP + l] = bv;
}

// ---------------- 3) attention helpers ----------------
static __device__ inline v8f wmma_bf16(v16bf a, v16bf b, v8f c) {
    return __builtin_amdgcn_wmma_f32_16x16x32_bf16(false, a, false, b, (short)0, c,
                                                   false, false);
}

// P·V over NT adjacent-owned output tiles: one LDS A-frag amortized over NT WMMAs.
template <int NT>
static __device__ inline void pv_tiles(const bf16t* __restrict__ S,
                                       const bf16t* __restrict__ qTb,
                                       const float* __restrict__ ri,
                                       float* __restrict__ yhw,
                                       int b, int rt, int n, int kh, int t0) {
    v8f zero = {};
    v8f acc[NT];
    const bf16t* vcol[NT];
    #pragma unroll
    for (int i = 0; i < NT; ++i) {
        acc[i] = zero;
        vcol[i] = qTb + ((size_t)(t0 + i) * 16 + n) * LP + kh * 8;
    }
    const bf16t* arow = S + n * LP + kh * 8;
    for (int mt = 0; mt < MTILES; ++mt) {
        v16bf af = ld_fragA(arow + mt * 32);
        v16bf bf[NT];
        #pragma unroll
        for (int i = 0; i < NT; ++i) bf[i] = ld_fragA(vcol[i] + mt * 32);
        #pragma unroll
        for (int i = 0; i < NT; ++i) acc[i] = wmma_bf16(af, bf[i], acc[i]);
    }
    #pragma unroll
    for (int i = 0; i < NT; ++i) {
        int col = (t0 + i) * 16 + n;
        #pragma unroll
        for (int r = 0; r < 8; ++r) {
            int M = r + kh * 8;
            int l = rt * 16 + M;
            if (l < LL && col < DD)
                yhw[((size_t)b * LL + l) * DD + col] = acc[i][r] * ri[r];
        }
    }
}

// ---------------- 3) attention: one WG (8 waves) per (b, 16-row tile) ----
__global__ __launch_bounds__(256) void k_attn(const bf16t* __restrict__ q,
                                              const bf16t* __restrict__ qT,
                                              float* __restrict__ yhw) {
    extern __shared__ char smem[];
    bf16t* S = (bf16t*)smem;                       // [16][LP] scores -> probs
    float* rowinv = (float*)(smem + 16 * LP * 2);  // [16]

    const int tid  = threadIdx.x;
    const int wv   = tid >> 5;
    const int lane = tid & 31;
    const int n    = lane & 15;     // A-row / B-col / C-col selector
    const int kh   = lane >> 4;     // half-wave selector
    const int b    = blockIdx.x / ROWTILES;
    const int rt   = blockIdx.x % ROWTILES;

    // Q row-tile as 10 A-fragments (16x32 bf16 each)
    v16bf aQ[10];
    {
        const bf16t* qrow = q + ((size_t)b * LP + rt * 16 + n) * DP + kh * 8;
        #pragma unroll
        for (int kb = 0; kb < 10; ++kb) aQ[kb] = ld_fragA(qrow + kb * 32);
    }

    // Phase A: S = 10 * Q K^T  (masked columns -> -inf), bf16 into LDS.
    // Load all 10 B fragments first (one clause), then 10 back-to-back WMMAs.
    const float NEGF = -__builtin_inff();
    for (int ct = wv; ct < COLTILES; ct += 8) {
        const bf16t* krow = q + ((size_t)b * LP + ct * 16 + n) * DP + kh * 8;
        v16bf vB[10];
        #pragma unroll
        for (int kb = 0; kb < 10; ++kb) vB[kb] = ld_fragA(krow + kb * 32);
        v8f acc = {};
        #pragma unroll
        for (int kb = 0; kb < 10; ++kb) acc = wmma_bf16(aQ[kb], vB[kb], acc);
        int mcol = ct * 16 + n;
        #pragma unroll
        for (int r = 0; r < 8; ++r) {
            int M = r + kh * 8;
            float v = (mcol < LL) ? acc[r] * SCL : NEGF;  // branchless select
            S[M * LP + ct * 16 + n] = f2bf(v);
        }
    }
    __syncthreads();

    // Phase B: row softmax (16 threads per row), rewrite P (bf16) in place
    {
        int r3 = tid >> 4, j = tid & 15;
        float mx = -__builtin_inff();
        for (int c = j; c < LP; c += 16) mx = fmaxf(mx, bf2f(S[r3 * LP + c]));
        mx = fmaxf(mx, __shfl_xor(mx, 1, 32));
        mx = fmaxf(mx, __shfl_xor(mx, 2, 32));
        mx = fmaxf(mx, __shfl_xor(mx, 4, 32));
        mx = fmaxf(mx, __shfl_xor(mx, 8, 32));
        float sum = 0.f;
        for (int c = j; c < LP; c += 16) {
            float p = __expf(bf2f(S[r3 * LP + c]) - mx);
            sum += p;
            S[r3 * LP + c] = f2bf(p);
        }
        sum += __shfl_xor(sum, 1, 32);
        sum += __shfl_xor(sum, 2, 32);
        sum += __shfl_xor(sum, 4, 32);
        sum += __shfl_xor(sum, 8, 32);
        if (j == 0) rowinv[r3] = 1.0f / sum;
    }
    __syncthreads();

    // Phase C: O = P V. Waves 0-3 own 3 tiles (cols 0..191), waves 4-7 own 2.
    float ri[8];
    #pragma unroll
    for (int r = 0; r < 8; ++r) ri[r] = rowinv[r + kh * 8];
    const bf16t* qTb = qT + (size_t)b * DP * LP;
    if (wv < 4) pv_tiles<3>(S, qTb, ri, yhw, b, rt, n, kh, wv * 3);
    else        pv_tiles<2>(S, qTb, ri, yhw, b, rt, n, kh, 12 + (wv - 4) * 2);
}

// ---------------- 4) per-patch 49x49 channel attention -> yc[b][l][p*6+c] ----
__global__ void k_chan(const bf16t* __restrict__ q, float* __restrict__ yc) {
    __shared__ float pc[49 * CC];
    int bl = blockIdx.x;
    int b = bl / LL, l = bl % LL;
    for (int i = threadIdx.x; i < DD; i += 64) {
        int c = i / 49, p = i % 49;
        pc[p * CC + c] = bf2f(q[((size_t)b * LP + l) * DP + i]);
    }
    __syncthreads();
    int p = threadIdx.x;
    if (p >= 49) return;
    float mx = -__builtin_inff();
    for (int qq = 0; qq < 49; ++qq) {
        float dot = 0.f;
        #pragma unroll
        for (int c = 0; c < CC; ++c) dot += pc[p * CC + c] * pc[qq * CC + c];
        mx = fmaxf(mx, dot * SCL);
    }
    float sum = 0.f;
    float o[CC] = {0.f, 0.f, 0.f, 0.f, 0.f, 0.f};
    for (int qq = 0; qq < 49; ++qq) {
        float dot = 0.f;
        #pragma unroll
        for (int c = 0; c < CC; ++c) dot += pc[p * CC + c] * pc[qq * CC + c];
        float e = __expf(dot * SCL - mx);
        sum += e;
        #pragma unroll
        for (int c = 0; c < CC; ++c) o[c] += e * pc[qq * CC + c];
    }
    float rinv = 1.0f / sum;
    #pragma unroll
    for (int c = 0; c < CC; ++c)
        yc[((size_t)b * LL + l) * DD + p * CC + c] = o[c] * rinv;
}

// ---------------- 5) fold(overlap-add gather) / mask + residual ----
__global__ void k_fold_combine(const float* __restrict__ xb,
                               const float* __restrict__ yhw,
                               const float* __restrict__ ycb,
                               float* __restrict__ out) {
    int idx = blockIdx.x * blockDim.x + threadIdx.x;
    if (idx >= BB * CC * HH * WW) return;
    int b = idx / (CC * HH * WW);
    int r = idx % (CC * HH * WW);
    int c = r / (HH * WW);
    int rr = r % (HH * WW);
    int y = rr / WW, x = rr % WW;

    int t = y + PADA, u = x + PADA;
    int iy[2], lh[2], ny = 0;
    int jx[2], lw[2], nx = 0;
    { int i0 = t & 3;          iy[ny] = i0; lh[ny] = (t - i0) >> 2; ++ny;
      int i1 = (t & 3) + 4;    if (i1 < KS && t >= i1) { iy[ny] = i1; lh[ny] = (t - i1) >> 2; ++ny; } }
    { int j0 = u & 3;          jx[nx] = j0; lw[nx] = (u - j0) >> 2; ++nx;
      int j1 = (u & 3) + 4;    if (j1 < KS && u >= j1) { jx[nx] = j1; lw[nx] = (u - j1) >> 2; ++nx; } }

    float acc = 0.f;
    for (int a = 0; a < ny; ++a)
        for (int e = 0; e < nx; ++e) {
            int l = lh[a] * LH + lw[e];
            int d = c * 49 + iy[a] * 7 + jx[e];
            size_t o = ((size_t)b * LL + l) * DD + d;
            acc += yhw[o] + ycb[o];
        }
    float cnt = (float)(ny * nx);
    out[idx] = xb[idx] + acc / cnt;
}

// ---------------- launch ----------------
extern "C" void kernel_launch(void* const* d_in, const int* in_sizes, int n_in,
                              void* d_out, int out_size, void* d_ws, size_t ws_size,
                              hipStream_t stream) {
    (void)in_sizes; (void)n_in; (void)out_size; (void)ws_size;
    const float* x  = (const float*)d_in[0];
    const float* w1 = (const float*)d_in[1];
    const float* w2 = (const float*)d_in[2];
    float* out = (float*)d_out;

    auto align_up = [](size_t v) { return (v + 255) & ~(size_t)255; };
    size_t off = 0;
    size_t xb_off = off;  off = align_up(off + (size_t)BB * CC * HH * WW * sizeof(float));
    size_t q_off  = off;  off = align_up(off + (size_t)BB * LP * DP * sizeof(bf16t));
    size_t qT_off = off;  off = align_up(off + (size_t)BB * DP * LP * sizeof(bf16t));
    size_t yh_off = off;  off = align_up(off + (size_t)BB * LL * DD * sizeof(float));
    size_t yc_off = off;  off = align_up(off + (size_t)BB * LL * DD * sizeof(float));

    char* ws = (char*)d_ws;
    float* xb  = (float*)(ws + xb_off);
    bf16t* q   = (bf16t*)(ws + q_off);
    bf16t* qT  = (bf16t*)(ws + qT_off);
    float* yhw = (float*)(ws + yh_off);
    float* ycb = (float*)(ws + yc_off);

    { int n = BB * HH * WW;
      k_basic_block<<<(n + 255) / 256, 256, 0, stream>>>(x, w1, w2, xb); }

    { size_t n = (size_t)BB * LP * DP;
      k_pack<<<(unsigned)((n + 255) / 256), 256, 0, stream>>>(xb, q, qT); }

    { unsigned grid = BB * ROWTILES;
      size_t smem = (size_t)16 * LP * sizeof(bf16t) + 16 * sizeof(float);
      k_attn<<<grid, 256, smem, stream>>>(q, qT, yhw); }

    { unsigned grid = BB * LL;
      k_chan<<<grid, 64, 0, stream>>>(q, ycb); }

    { int n = BB * CC * HH * WW;
      k_fold_combine<<<(n + 255) / 256, 256, 0, stream>>>(xb, yhw, ycb, out); }
}
